// PerformerModel_40604620817009
// MI455X (gfx1250) — matrix operand order repaired and checked by reading, not compiled, not run on hardware
//
#include <hip/hip_runtime.h>
#include <hip/hip_bf16.h>

typedef _Float16 f16;
typedef __attribute__((ext_vector_type(16))) _Float16 v16h;
typedef __attribute__((ext_vector_type(8)))  _Float16 v8h;
typedef __attribute__((ext_vector_type(4)))  _Float16 v4h;
typedef __attribute__((ext_vector_type(8)))  float    v8f;

// ---------------- model constants ----------------
constexpr int CL = 4, CH = 4, CDH = 32, CM = 110, CMP = 128, CD = 128, CFF = 512;
constexpr int CB = 64, CS = 800, CIN = 4;
constexpr long long BS = (long long)CB * CS;   // 51200 tokens
constexpr int BH = CB * CH;                    // 256 (b,h) pairs
constexpr float DN     = 0.42044820762685725f;  // 32^-0.25
constexpr float DNSQH  = 0.08838834764831845f;  // 0.5 * 32^-0.5
constexpr float RATIO  = 0.09534625892455924f;  // 110^-0.5
constexpr float KEPS   = 1e-4f;

// ---------------- wave reductions (wave32) ----------------
__device__ inline float wave_sum(float v) {
#pragma unroll
  for (int o = 16; o > 0; o >>= 1) v += __shfl_xor(v, o, 32);
  return v;
}
__device__ inline float wave_max(float v) {
#pragma unroll
  for (int o = 16; o > 0; o >>= 1) v = fmaxf(v, __shfl_xor(v, o, 32));
  return v;
}

// ---------------- generic batched WMMA GEMM ----------------
// Tile: BM=128 x BN=64 x BK=32. 8 waves; wave (wm,wn) owns 32 rows x 32 cols
// = 4 x v_wmma_f32_16x16x32_f16 per K-step. LDS double-buffered; B staged
// transposed so both A and B fragments are contiguous ds_load_b128 pairs.
//
// CONTRACT (all call sites satisfy this; lets staging be branch-light):
//   Kd % 32 == 0, Nc % 8 == 0, and if TRA then Mr % 8 == 0.
constexpr int BM = 128, BN = 64, BK = 32;
constexpr int LDSA = BK + 8;  // 40 f16 row stride (80B, keeps 16B chunk alignment)
constexpr int LDSB = BK + 8;

struct GemmP {
  const f16* A; const f16* Bm;
  float* Cf; f16* Ch;
  const float* bias; const float* resid; const float* rowscale;
  long long sAo, sAi, sBo, sBi, sCo, sCi;
  int lda, ldb, ldc;
  int Mr, Nc, Kd;
  int innerN;      // inner batch count (e.g. H); z = zo*innerN + zi
  int rsStride;    // rowscale stride per z
};

// EPI: 0=f32 store, 1=f16 store, 2=bias+resid->f32, 3=bias+GELU->f16, 4=rowscale->f16
template <int EPI, bool TRA>
__global__ __launch_bounds__(256) void wmma_gemm(GemmP p) {
  __shared__ f16 As[2][BM][LDSA];
  __shared__ f16 Bs[2][BN][LDSB];   // transposed: Bs[n][k]
  const int tid  = threadIdx.x;
  const int lane = tid & 31;
  const int wid  = tid >> 5;
  const int wm   = wid >> 1;   // 0..3 : 32-row strip
  const int wn   = wid & 1;    // 0..1 : 32-col strip
  const int lrow = lane & 15;
  const int lhi  = lane >> 4;

  const int z  = blockIdx.z;
  const int zo = z / p.innerN, zi = z % p.innerN;
  const long long offA = (long long)zo * p.sAo + (long long)zi * p.sAi;
  const long long offB = (long long)zo * p.sBo + (long long)zi * p.sBi;
  const long long offC = (long long)zo * p.sCo + (long long)zi * p.sCi;

  const int rowBase = blockIdx.y * BM;
  const int colBase = blockIdx.x * BN;
  const int nk = p.Kd / BK;

  v8h ra[2];  // staged A chunks (2 x 8 f16 per thread)
  v8h rb;     // staged B chunk  (1 x 8 f16 per thread)

  // ---- staging: global -> regs (chunk-level predication only) ----
  auto loadA = [&](int kt) {
    const int k0 = kt * BK;
#pragma unroll
    for (int c = 0; c < 2; ++c) {
      const int q = c * 256 + tid;   // chunk id in [0,512)
      v8h v = {};
      if (!TRA) {
        const int r = q >> 2, kc = (q & 3) << 3;      // row, k-chunk
        const int gr = rowBase + r;
        if (gr < p.Mr)
          v = *(const v8h*)&p.A[offA + (long long)gr * p.lda + (k0 + kc)];
      } else {
        const int kk = q & 31, r0 = (q >> 5) << 3;    // k, row-chunk
        const int gr = rowBase + r0;
        if (gr + 8 <= p.Mr)
          v = *(const v8h*)&p.A[offA + (long long)(k0 + kk) * p.lda + gr];
      }
      ra[c] = v;
    }
  };
  auto loadB = [&](int kt) {
    const int k0 = kt * BK;
    const int kk = tid >> 3, n0 = (tid & 7) << 3;
    const int gn = colBase + n0;
    v8h v = {};
    if (gn + 8 <= p.Nc)
      v = *(const v8h*)&p.Bm[offB + (long long)(k0 + kk) * p.ldb + gn];
    rb = v;
  };
  // ---- staging: regs -> LDS ----
  auto storeA = [&](int buf) {
#pragma unroll
    for (int c = 0; c < 2; ++c) {
      const int q = c * 256 + tid;
      if (!TRA) {
        const int r = q >> 2, kc = (q & 3) << 3;
        *(v8h*)&As[buf][r][kc] = ra[c];               // ds_store_b128
      } else {
        const int kk = q & 31, r0 = (q >> 5) << 3;
#pragma unroll
        for (int i = 0; i < 8; ++i) As[buf][r0 + i][kk] = ra[c][i];
      }
    }
  };
  auto storeB = [&](int buf) {
    const int kk = tid >> 3, n0 = (tid & 7) << 3;
#pragma unroll
    for (int i = 0; i < 8; ++i) Bs[buf][n0 + i][kk] = rb[i];  // transpose scatter
  };

  v8f acc[2][2];
#pragma unroll
  for (int ta = 0; ta < 2; ++ta)
#pragma unroll
    for (int t = 0; t < 2; ++t) acc[ta][t] = {};

  auto compute = [&](int buf) {
    v16h bfr[2];
#pragma unroll
    for (int t = 0; t < 2; ++t) {
      const int n = wn * 32 + t * 16 + lrow;
      v8h b0 = *(const v8h*)&Bs[buf][n][lhi * 8];         // ds_load_b128
      v8h b1 = *(const v8h*)&Bs[buf][n][16 + lhi * 8];
      bfr[t] = __builtin_shufflevector(b0, b1, 0, 1, 2, 3, 4, 5, 6, 7,
                                       8, 9, 10, 11, 12, 13, 14, 15);
    }
#pragma unroll
    for (int ta = 0; ta < 2; ++ta) {
      const int arow = wm * 32 + ta * 16 + lrow;
      v8h a0 = *(const v8h*)&As[buf][arow][lhi * 8];       // ds_load_b128
      v8h a1 = *(const v8h*)&As[buf][arow][16 + lhi * 8];
      v16h af = __builtin_shufflevector(a0, a1, 0, 1, 2, 3, 4, 5, 6, 7,
                                        8, 9, 10, 11, 12, 13, 14, 15);
#pragma unroll
      for (int t = 0; t < 2; ++t)
        acc[ta][t] = __builtin_amdgcn_wmma_f32_16x16x32_f16(
            false, af, false, bfr[t], (short)0, acc[ta][t], false, false);
    }
  };

  // ---- main loop: register double-buffering over LDS ping-pong ----
  loadA(0); loadB(0);
  storeA(0); storeB(0);
  for (int kt = 0; kt < nk; ++kt) {
    const bool more = (kt + 1 < nk);
    if (more) { loadA(kt + 1); loadB(kt + 1); }  // global loads in flight
    __syncthreads();                             // buf (kt&1) ready
    compute(kt & 1);
    if (more) { storeA((kt + 1) & 1); storeB((kt + 1) & 1); }
  }

  // ---- epilogue: f32 C layout -> lane n = lane&15, m = r + 8*(lane>>4) ----
  // Pointer-stepped stores (one 64-bit address per tile, then += ldc per row).
#pragma unroll
  for (int ta = 0; ta < 2; ++ta) {
    const int gm0  = rowBase + wm * 32 + ta * 16 + lhi * 8;
    const int rmax = p.Mr - gm0;   // rows this lane may write (<=0: none)
#pragma unroll
    for (int t = 0; t < 2; ++t) {
      const int gn = colBase + wn * 32 + t * 16 + lrow;
      if (gn < p.Nc) {
        const long long ci0 = offC + (long long)gm0 * p.ldc + gn;
        if (EPI == 0 || EPI == 2) {
          float* cp = p.Cf + ci0;
          const float* rp = (EPI == 2) ? (p.resid + ci0) : nullptr;
          const float bb = (EPI == 2) ? p.bias[gn] : 0.f;
#pragma unroll
          for (int r = 0; r < 8; ++r) {
            if (r < rmax) {
              if (EPI == 0) *cp = acc[ta][t][r];
              else          *cp = *rp + bb + acc[ta][t][r];
            }
            cp += p.ldc;
            if (EPI == 2) rp += p.ldc;
          }
        } else {
          f16* cp = p.Ch + ci0;
          const float* rs = (EPI == 4)
              ? (p.rowscale + (long long)z * p.rsStride + gm0) : nullptr;
          const float bb = (EPI == 3) ? p.bias[gn] : 0.f;
#pragma unroll
          for (int r = 0; r < 8; ++r) {
            if (r < rmax) {
              float v = acc[ta][t][r];
              if (EPI == 1) {
                *cp = (f16)v;
              } else if (EPI == 3) {
                float xg = v + bb;
                *cp = (f16)(0.5f * xg * (1.f + erff(xg * 0.70710678118654752f)));
              } else {  // EPI == 4
                *cp = (f16)(v * rs[r]);
              }
            }
            cp += p.ldc;
          }
        }
      }
    }
  }
}

// ---------------- small kernels ----------------
__global__ void embed_kernel(const float* x, const float* embW, const float* embb,
                             const float* pos, float* h) {
  long long idx = (long long)blockIdx.x * blockDim.x + threadIdx.x;
  if (idx >= BS * CD) return;
  int d = (int)(idx % CD);
  long long bs = idx / CD;
  int s = (int)(bs % CS);
  const float* xr = x + bs * CIN;
  float acc = embb[d] + pos[(long long)s * CD + d];
#pragma unroll
  for (int i = 0; i < CIN; ++i) acc += xr[i] * embW[i * CD + d];
  h[idx] = acc;
}

// vectorized fp32 -> f16 cast (n % 4 == 0)
__global__ void cast_f16_kernel(const float* src, f16* dst, long long n4) {
  long long i4 = (long long)blockIdx.x * blockDim.x + threadIdx.x;
  if (i4 >= n4) return;
  float4 s4 = *(const float4*)&src[i4 * 4];
  v4h d4 = {(f16)s4.x, (f16)s4.y, (f16)s4.z, (f16)s4.w};
  *(v4h*)&dst[i4 * 4] = d4;
}

// proj [L,M,DH] -> projT [L,DH,MP] scaled by dn, zero-padded cols
__global__ void proj_prep_kernel(const float* proj, f16* projT) {
  int idx = blockIdx.x * blockDim.x + threadIdx.x;
  if (idx >= CL * CDH * CMP) return;
  int m = idx % CMP;
  int kk = (idx / CMP) % CDH;
  int l = idx / (CMP * CDH);
  float v = (m < CM) ? proj[((long long)l * CM + m) * CDH + kk] * DN : 0.f;
  projT[idx] = (f16)v;
}

// LayerNorm over D=128, one wave per row, f16 output
__global__ __launch_bounds__(256) void ln_kernel(const float* h, const float* sc,
                                                 const float* bi, f16* outp) {
  int wid = threadIdx.x >> 5, lane = threadIdx.x & 31;
  long long row = (long long)blockIdx.x * 8 + wid;
  if (row >= BS) return;
  float4 t = *(const float4*)&h[row * CD + lane * 4];
  float v[4] = {t.x, t.y, t.z, t.w};
  float sum = v[0] + v[1] + v[2] + v[3];
  sum = wave_sum(sum);
  float mu = sum * (1.f / CD);
  float var = 0.f;
#pragma unroll
  for (int i = 0; i < 4; ++i) { float d = v[i] - mu; var += d * d; }
  var = wave_sum(var) * (1.f / CD);
  float rs = rsqrtf(var + 1e-5f);
  float4 s4 = *(const float4*)&sc[lane * 4];
  float4 b4 = *(const float4*)&bi[lane * 4];
  v4h o = {(f16)((v[0] - mu) * rs * s4.x + b4.x),
           (f16)((v[1] - mu) * rs * s4.y + b4.y),
           (f16)((v[2] - mu) * rs * s4.z + b4.z),
           (f16)((v[3] - mu) * rs * s4.w + b4.w)};
  *(v4h*)&outp[row * CD + lane * 4] = o;
}

// FAVOR+ exp: one wave per (b,h,s) row of dd [B,H,S,MP]
template <bool ISQ>
__global__ __launch_bounds__(256) void favor_exp_kernel(const float* dd, const f16* qk,
                                                        const float* gmax, f16* outp) {
  int wid = threadIdx.x >> 5, lane = threadIdx.x & 31;
  long long row = (long long)blockIdx.x * 8 + wid;
  if (row >= (long long)BH * CS) return;
  int s  = (int)(row % CS);
  int bh = (int)(row / CS);
  int hh = bh % CH, bb = bh / CH;
  float4 d4 = *(const float4*)&dd[row * CMP + lane * 4];
  float dv[4] = {d4.x, d4.y, d4.z, d4.w};
#pragma unroll
  for (int i = 0; i < 4; ++i)
    if (lane * 4 + i >= CM) dv[i] = -1e30f;
  float mx;
  if (ISQ) {
    float m = fmaxf(fmaxf(dv[0], dv[1]), fmaxf(dv[2], dv[3]));
    mx = wave_max(m);
  } else {
    mx = gmax[0];
  }
  float qv = (float)qk[((long long)bb * CS + s) * CD + hh * CDH + lane];
  float diag = DNSQH * wave_sum(qv * qv);
  v4h o;
#pragma unroll
  for (int i = 0; i < 4; ++i) {
    int c = lane * 4 + i;
    o[i] = (c < CM) ? (f16)(RATIO * (expf(dv[i] - diag - mx) + KEPS)) : (f16)0.f;
  }
  *(v4h*)&outp[row * CMP + lane * 4] = o;
}

constexpr int RED_BLOCKS = 1024;
__global__ __launch_bounds__(256) void maxred1_kernel(const float* dd, float* partial,
                                                      long long n4) {
  __shared__ float sred[256];
  float m = -1e30f;
  for (long long i4 = (long long)blockIdx.x * 256 + threadIdx.x; i4 < n4;
       i4 += (long long)gridDim.x * 256) {
    float4 v = *(const float4*)&dd[i4 * 4];
    int c = (int)((i4 * 4) & (CMP - 1));
    if (c + 0 < CM) m = fmaxf(m, v.x);
    if (c + 1 < CM) m = fmaxf(m, v.y);
    if (c + 2 < CM) m = fmaxf(m, v.z);
    if (c + 3 < CM) m = fmaxf(m, v.w);
  }
  sred[threadIdx.x] = m; __syncthreads();
  for (int st = 128; st > 0; st >>= 1) {
    if (threadIdx.x < st) sred[threadIdx.x] = fmaxf(sred[threadIdx.x], sred[threadIdx.x + st]);
    __syncthreads();
  }
  if (threadIdx.x == 0) partial[blockIdx.x] = sred[0];
}
__global__ __launch_bounds__(256) void maxred2_kernel(const float* partial, float* gmax) {
  __shared__ float sred[256];
  float m = -1e30f;
  for (int i = threadIdx.x; i < RED_BLOCKS; i += 256) m = fmaxf(m, partial[i]);
  sred[threadIdx.x] = m; __syncthreads();
  for (int st = 128; st > 0; st >>= 1) {
    if (threadIdx.x < st) sred[threadIdx.x] = fmaxf(sred[threadIdx.x], sred[threadIdx.x + st]);
    __syncthreads();
  }
  if (threadIdx.x == 0) gmax[0] = sred[0];
}

// ksum[b,h,m] = sum_s kp[b,h,s,m]; one thread per 8-wide m chunk
__global__ void ksum_kernel(const f16* kp, float* ksum) {
  int idx = blockIdx.x * blockDim.x + threadIdx.x;
  if (idx >= BH * (CMP / 8)) return;
  int mc = (idx % (CMP / 8)) * 8;
  int bh = idx / (CMP / 8);
  const f16* p = kp + (long long)bh * CS * CMP + mc;
  float acc[8] = {};
  for (int s = 0; s < CS; ++s) {
    v8h v = *(const v8h*)&p[(long long)s * CMP];
#pragma unroll
    for (int i = 0; i < 8; ++i) acc[i] += (float)v[i];
  }
#pragma unroll
  for (int i = 0; i < 8; ++i) ksum[bh * CMP + mc + i] = acc[i];
}

// d_inv[b,h,s] = 1 / (qp[b,h,s,:] . ksum[b,h,:]); one wave per row
__global__ __launch_bounds__(256) void dinv_kernel(const f16* qp, const float* ksum,
                                                   float* dinv) {
  int wid = threadIdx.x >> 5, lane = threadIdx.x & 31;
  long long row = (long long)blockIdx.x * 8 + wid;
  if (row >= (long long)BH * CS) return;
  int bh = (int)(row / CS);
  v4h qv = *(const v4h*)&qp[row * CMP + lane * 4];
  float4 kv = *(const float4*)&ksum[bh * CMP + lane * 4];
  float acc = (float)qv[0] * kv.x + (float)qv[1] * kv.y +
              (float)qv[2] * kv.z + (float)qv[3] * kv.w;
  acc = wave_sum(acc);
  if (lane == 0) dinv[row] = 1.f / acc;
}

__global__ void mean_kernel(const float* h, float* hm) {
  int idx = blockIdx.x * blockDim.x + threadIdx.x;
  if (idx >= CB * CD) return;
  int d = idx % CD, b = idx / CD;
  const float* p = h + ((long long)b * CS) * CD + d;
  float acc = 0.f;
  for (int s = 0; s < CS; ++s) acc += p[(long long)s * CD];
  hm[idx] = acc * (1.f / CS);
}

__global__ void fc_kernel(const float* hm, const float* fcW, const float* fcb, float* outp) {
  int idx = blockIdx.x * blockDim.x + threadIdx.x;
  if (idx >= CB * 2) return;
  int c = idx % 2, b = idx / 2;
  float acc = fcb[c];
  for (int d = 0; d < CD; ++d) acc += hm[b * CD + d] * fcW[d * 2 + c];
  outp[idx] = acc;
}

// ---------------- host-side GEMM dispatch ----------------
static void launch_gemm(hipStream_t st, int epi, bool tra, const GemmP& p, int batch) {
  dim3 g((p.Nc + BN - 1) / BN, (p.Mr + BM - 1) / BM, batch);
  dim3 b(256);
  if (!tra) {
    switch (epi) {
      case 0: wmma_gemm<0, false><<<g, b, 0, st>>>(p); break;
      case 1: wmma_gemm<1, false><<<g, b, 0, st>>>(p); break;
      case 2: wmma_gemm<2, false><<<g, b, 0, st>>>(p); break;
      case 3: wmma_gemm<3, false><<<g, b, 0, st>>>(p); break;
      case 4: wmma_gemm<4, false><<<g, b, 0, st>>>(p); break;
    }
  } else {
    wmma_gemm<1, true><<<g, b, 0, st>>>(p);  // only f16-store TRA variant is used
  }
}

extern "C" void kernel_launch(void* const* d_in, const int* in_sizes, int n_in,
                              void* d_out, int out_size, void* d_ws, size_t ws_size,
                              hipStream_t stream) {
  const float* x    = (const float*)d_in[0];
  const float* embW = (const float*)d_in[1];
  const float* embb = (const float*)d_in[2];
  const float* pos  = (const float*)d_in[3];
  const float* ln1s = (const float*)d_in[4];
  const float* ln1b = (const float*)d_in[5];
  const float* Wq   = (const float*)d_in[6];
  const float* Wk   = (const float*)d_in[7];
  const float* Wv   = (const float*)d_in[8];
  const float* Wo   = (const float*)d_in[9];
  const float* bo   = (const float*)d_in[10];
  const float* proj = (const float*)d_in[11];
  const float* ln2s = (const float*)d_in[12];
  const float* ln2b = (const float*)d_in[13];
  const float* W1   = (const float*)d_in[14];
  const float* b1   = (const float*)d_in[15];
  const float* W2   = (const float*)d_in[16];
  const float* b2   = (const float*)d_in[17];
  const float* fcW  = (const float*)d_in[18];
  const float* fcb  = (const float*)d_in[19];
  float* outp = (float*)d_out;

  // -------- workspace carving --------
  char* w = (char*)d_ws;
  size_t off = 0;
  auto alloc = [&](size_t bytes) -> void* {
    void* p = w + off;
    off += (bytes + 255) & ~(size_t)255;
    return p;
  };
  float* h    = (float*)alloc(BS * CD * 4);
  f16*   aH   = (f16*)alloc(BS * CD * 2);
  f16*   qH   = (f16*)alloc(BS * CD * 2);
  f16*   kH   = (f16*)alloc(BS * CD * 2);
  f16*   vH   = (f16*)alloc(BS * CD * 2);
  float* dd   = (float*)alloc((long long)BH * CS * CMP * 4);
  f16*   gH   = (f16*)dd;  // FFN intermediate aliases dd (dd dead by then)
  f16*   qp   = (f16*)alloc((long long)BH * CS * CMP * 2);
  f16*   kp   = (f16*)alloc((long long)BH * CS * CMP * 2);
  f16*   ctx  = (f16*)alloc((long long)BH * CMP * CDH * 2);
  f16*   oH   = (f16*)alloc(BS * CD * 2);
  float* ksum = (float*)alloc(BH * CMP * 4);
  float* dinv = (float*)alloc((long long)BH * CS * 4);
  float* hm   = (float*)alloc(CB * CD * 4);
  float* gmax = (float*)alloc(256);
  float* part = (float*)alloc(RED_BLOCKS * 4);
  f16*   WqH  = (f16*)alloc((long long)CL * CD * CD * 2);
  f16*   WkH  = (f16*)alloc((long long)CL * CD * CD * 2);
  f16*   WvH  = (f16*)alloc((long long)CL * CD * CD * 2);
  f16*   WoH  = (f16*)alloc((long long)CL * CD * CD * 2);
  f16*   W1H  = (f16*)alloc((long long)CL * CD * CFF * 2);
  f16*   W2H  = (f16*)alloc((long long)CL * CFF * CD * 2);
  f16*   pT   = (f16*)alloc((long long)CL * CDH * CMP * 2);

  // -------- weight preparation --------
  auto cast = [&](const float* s, f16* d, long long n) {
    long long n4 = n / 4;
    cast_f16_kernel<<<(unsigned)((n4 + 255) / 256), 256, 0, stream>>>(s, d, n4);
  };
  cast(Wq, WqH, (long long)CL * CD * CD);
  cast(Wk, WkH, (long long)CL * CD * CD);
  cast(Wv, WvH, (long long)CL * CD * CD);
  cast(Wo, WoH, (long long)CL * CD * CD);
  cast(W1, W1H, (long long)CL * CD * CFF);
  cast(W2, W2H, (long long)CL * CFF * CD);
  proj_prep_kernel<<<(CL * CDH * CMP + 255) / 256, 256, 0, stream>>>(proj, pT);

  // -------- embedding + positional --------
  embed_kernel<<<(unsigned)((BS * CD + 255) / 256), 256, 0, stream>>>(x, embW, embb, pos, h);

  const int rowBlks = (int)((BS + 7) / 8);            // ln rows
  const int bhsBlks = (int)(((long long)BH * CS + 7) / 8);

  for (int l = 0; l < CL; ++l) {
    // ---- LN1 -> aH (f16) ----
    ln_kernel<<<rowBlks, 256, 0, stream>>>(h, ln1s + l * CD, ln1b + l * CD, aH);

    // ---- q,k,v = aH @ W (f16 out) ----
    for (int j = 0; j < 3; ++j) {
      GemmP p{};
      p.A = aH; p.Bm = (j == 0 ? WqH : j == 1 ? WkH : WvH) + (long long)l * CD * CD;
      p.Ch = (j == 0 ? qH : j == 1 ? kH : vH);
      p.lda = CD; p.ldb = CD; p.ldc = CD;
      p.Mr = (int)BS; p.Nc = CD; p.Kd = CD; p.innerN = 1;
      launch_gemm(stream, 1, false, p, 1);
    }

    // ---- dd_q = q @ projT  (batched over b,h; f32 out [B,H,S,MP]) ----
    {
      GemmP p{};
      p.A = qH; p.Bm = pT + (long long)l * CDH * CMP; p.Cf = dd;
      p.lda = CD; p.ldb = CMP; p.ldc = CMP;
      p.sAo = (long long)CS * CD; p.sAi = CDH;
      p.sCo = (long long)CH * CS * CMP; p.sCi = (long long)CS * CMP;
      p.Mr = CS; p.Nc = CMP; p.Kd = CDH; p.innerN = CH;
      launch_gemm(stream, 0, false, p, BH);
    }
    favor_exp_kernel<true><<<bhsBlks, 256, 0, stream>>>(dd, qH, gmax, qp);

    // ---- dd_k = k @ projT ; global max ; kp ----
    {
      GemmP p{};
      p.A = kH; p.Bm = pT + (long long)l * CDH * CMP; p.Cf = dd;
      p.lda = CD; p.ldb = CMP; p.ldc = CMP;
      p.sAo = (long long)CS * CD; p.sAi = CDH;
      p.sCo = (long long)CH * CS * CMP; p.sCi = (long long)CS * CMP;
      p.Mr = CS; p.Nc = CMP; p.Kd = CDH; p.innerN = CH;
      launch_gemm(stream, 0, false, p, BH);
    }
    maxred1_kernel<<<RED_BLOCKS, 256, 0, stream>>>(dd, part, (long long)BH * CS * CMP / 4);
    maxred2_kernel<<<1, 256, 0, stream>>>(part, gmax);
    favor_exp_kernel<false><<<bhsBlks, 256, 0, stream>>>(dd, kH, gmax, kp);

    // ---- ksum, d_inv ----
    ksum_kernel<<<(BH * (CMP / 8) + 255) / 256, 256, 0, stream>>>(kp, ksum);
    dinv_kernel<<<bhsBlks, 256, 0, stream>>>(qp, ksum, dinv);

    // ---- ctx = kp^T @ v  (TRANSA; K=S) ----
    {
      GemmP p{};
      p.A = kp; p.Bm = vH; p.Ch = ctx;
      p.lda = CMP; p.ldb = CD; p.ldc = CDH;
      p.sAo = (long long)CH * CS * CMP; p.sAi = (long long)CS * CMP;
      p.sBo = (long long)CS * CD; p.sBi = CDH;
      p.sCo = (long long)CH * CMP * CDH; p.sCi = (long long)CMP * CDH;
      p.Mr = CMP; p.Nc = CDH; p.Kd = CS; p.innerN = CH;
      launch_gemm(stream, 1, true, p, BH);
    }

    // ---- o = (qp @ ctx) * d_inv  -> oH [B,S,H*DH] f16 ----
    {
      GemmP p{};
      p.A = qp; p.Bm = ctx; p.Ch = oH; p.rowscale = dinv;
      p.lda = CMP; p.ldb = CDH; p.ldc = CD;
      p.sAo = (long long)CH * CS * CMP; p.sAi = (long long)CS * CMP;
      p.sBo = (long long)CH * CMP * CDH; p.sBi = (long long)CMP * CDH;
      p.sCo = (long long)CS * CD; p.sCi = CDH;
      p.Mr = CS; p.Nc = CDH; p.Kd = CMP; p.innerN = CH; p.rsStride = CS;
      launch_gemm(stream, 4, false, p, BH);
    }

    // ---- h = h + o @ Wo + bo ----
    {
      GemmP p{};
      p.A = oH; p.Bm = WoH + (long long)l * CD * CD; p.Cf = h;
      p.resid = h; p.bias = bo + l * CD;
      p.lda = CD; p.ldb = CD; p.ldc = CD;
      p.Mr = (int)BS; p.Nc = CD; p.Kd = CD; p.innerN = 1;
      launch_gemm(stream, 2, false, p, 1);
    }

    // ---- LN2 -> aH ----
    ln_kernel<<<rowBlks, 256, 0, stream>>>(h, ln2s + l * CD, ln2b + l * CD, aH);

    // ---- g = gelu(aH @ W1 + b1) (f16) ----
    {
      GemmP p{};
      p.A = aH; p.Bm = W1H + (long long)l * CD * CFF; p.Ch = gH;
      p.bias = b1 + l * CFF;
      p.lda = CD; p.ldb = CFF; p.ldc = CFF;
      p.Mr = (int)BS; p.Nc = CFF; p.Kd = CD; p.innerN = 1;
      launch_gemm(stream, 3, false, p, 1);
    }

    // ---- h = h + g @ W2 + b2 ----
    {
      GemmP p{};
      p.A = gH; p.Bm = W2H + (long long)l * CFF * CD; p.Cf = h;
      p.resid = h; p.bias = b2 + l * CD;
      p.lda = CFF; p.ldb = CD; p.ldc = CD;
      p.Mr = (int)BS; p.Nc = CD; p.Kd = CFF; p.innerN = 1;
      launch_gemm(stream, 2, false, p, 1);
    }
  }

  // ---- mean pool + final FC ----
  mean_kernel<<<(CB * CD + 255) / 256, 256, 0, stream>>>(h, hm);
  fc_kernel<<<1, 128, 0, stream>>>(hm, fcW, fcb, outp);
}
